// PointNet2SemSegMSG_86852828659924
// MI455X (gfx1250) — matrix operand order repaired and checked by reading, not compiled, not run on hardware
//
#include <hip/hip_runtime.h>
#include <cstdint>
#include <cstddef>

typedef _Float16 h16;
typedef __attribute__((ext_vector_type(16))) _Float16 v16h;
typedef __attribute__((ext_vector_type(8)))  float    v8f;

#define DEV static __device__ __forceinline__

static const int BATCH = 8;
static const int NPTS  = 4096;

DEV float relu_f(float x) { return x > 0.f ? x : 0.f; }

// ---------------- WMMA fragment helpers (CDNA5 16x16x32 f16 layouts) ----------
// A tile in LDS: row-major [16 rows (m)][32 cols (k)] halves.
DEV v16h frag_A(const h16* At, int lane) {
  v16h a;
  int m = lane & 15;
  int h8 = (lane >> 4) << 4;
#pragma unroll
  for (int j = 0; j < 16; ++j) {
    int kk = j + h8;
    a[j] = At[m * 32 + kk];
  }
  return a;
}
// B tile in LDS: row-major [32 rows (k)][16 cols (n)] halves, row stride 16.
DEV v16h frag_B(const h16* Bt, int lane) {
  v16h b;
  int n = lane & 15;
  int h16v = (lane >> 4) << 4;
#pragma unroll
  for (int j = 0; j < 16; ++j) b[j] = Bt[(j + h16v) * 16 + n];
  return b;
}
// Stage A tile (16x32) from global W[Opad][Kp] into LDS (per-wave, 32 lanes).
DEV void stage_A(h16* dst, const h16* W, int Kp, int ro, int k0, int lane) {
  int r = lane & 15;
  int c = (lane >> 4) << 4;
  const uint4* s = (const uint4*)(W + (size_t)(ro + r) * Kp + k0 + c);
  uint4* d = (uint4*)(dst + r * 32 + c);
  d[0] = s[0]; d[1] = s[1];
}
// Stage B tile (32x16) from global X[Kp][P] into LDS (per-wave, 32 lanes).
DEV void stage_B(h16* dst, const h16* X, int P, int k0, int p0, int lane) {
  const uint4* s = (const uint4*)(X + (size_t)(k0 + lane) * P + p0);
  uint4* d = (uint4*)(dst + lane * 16);
  d[0] = s[0]; d[1] = s[1];
}

// ---------------- generic WMMA GEMM:  Y = relu(W @ X + bias) -----------------
// W: f16 [Opad][Kp]; X: f16 [B][Kp][P]; optional f32 out, f16 out, column sums.
__global__ void __launch_bounds__(128)
wmma_gemm_kernel(const h16* __restrict__ W, const float* __restrict__ bias,
                 const h16* __restrict__ X, int Kp, int P, int Opad, int Oreal,
                 float* __restrict__ Yf32, h16* __restrict__ Yf16,
                 float* __restrict__ colsum)
{
  __shared__ __align__(16) h16 At[4][16 * 32];
  __shared__ __align__(16) h16 Bt[4][32 * 16];
  int b = blockIdx.z;
  int wave = threadIdx.x >> 5, lane = threadIdx.x & 31;
  int p0 = blockIdx.x << 4;
  int ro = ((blockIdx.y << 2) + wave) << 4;
  if (ro >= Opad) return;            // whole-wave exit: EXEC stays all-ones for WMMA
  const h16* Xb = X + (size_t)b * Kp * P;
  v8f acc = {};
  int r = lane & 15;
  for (int k0 = 0; k0 < Kp; k0 += 32) {
    if (k0 + 32 < Kp) __builtin_prefetch(W + (size_t)(ro + r) * Kp + k0 + 32, 0, 1);
    stage_A(At[wave], W, Kp, ro, k0, lane);
    stage_B(Bt[wave], Xb, P, k0, p0, lane);
    asm volatile("s_wait_dscnt 0" ::: "memory");   // LDS stores visible to this wave
    v16h a  = frag_A(At[wave], lane);
    v16h bb = frag_B(Bt[wave], lane);
    acc = __builtin_amdgcn_wmma_f32_16x16x32_f16(false, a, false, bb, (short)0, acc,
                                                 false, false);
    asm volatile("s_wait_dscnt 0" ::: "memory");
  }
  int n = lane & 15, hf = lane >> 4;
  int p = p0 + n;
  float v[8];
#pragma unroll
  for (int i = 0; i < 8; ++i) {
    int o = ro + i + (hf << 3);
    v[i] = relu_f(acc[i] + bias[o]);
  }
  if (colsum) {
    float s[8];
#pragma unroll
    for (int i = 0; i < 8; ++i) s[i] = v[i];
#pragma unroll
    for (int m = 1; m < 16; m <<= 1)
#pragma unroll
      for (int i = 0; i < 8; ++i) s[i] += __shfl_xor(s[i], m, 32);
    if (n == 0) {
#pragma unroll
      for (int i = 0; i < 8; ++i) {
        int o = ro + i + (hf << 3);
        atomicAdd(&colsum[b * 448 + o], s[i]);
      }
    }
  }
#pragma unroll
  for (int i = 0; i < 8; ++i) {
    int o = ro + i + (hf << 3);
    if (o < Oreal) {
      if (Yf32) Yf32[((size_t)b * Oreal + o) * P + p] = v[i];
      if (Yf16) Yf16[((size_t)b * Opad + o) * P + p] = (h16)v[i];
    }
  }
}

// ---- fused pass-2: recompute h(448) tile in LDS, scale by attention, fuse GEMM,
// ---- relu, max-pool over k.  One block per (s, b); 8 waves.
__global__ void __launch_bounds__(256)
fuse_max_kernel(const h16* __restrict__ X, int Kp, int P, int kk,
                const h16* __restrict__ WB, const float* __restrict__ biasB,
                const float* __restrict__ Watt,
                const h16* __restrict__ WF, const float* __restrict__ biasF, int cout,
                float* __restrict__ out, int rowOff, int outRows, int S)
{
  __shared__ __align__(16) h16 Hs[448 * 16];
  __shared__ __align__(16) h16 Ast[8][16 * 32];
  __shared__ __align__(16) h16 Bst[8][32 * 16];
  int b = blockIdx.y, s = blockIdx.x;
  int wave = threadIdx.x >> 5, lane = threadIdx.x & 31;
  int n = lane & 15, hf = lane >> 4;
  const h16* Xb = X + (size_t)b * Kp * P;
  const float* wa = Watt + b * 448;
  int Ot = cout >> 4;
  float rm[2][8];
#pragma unroll
  for (int q = 0; q < 2; ++q)
#pragma unroll
    for (int i = 0; i < 8; ++i) rm[q][i] = 0.f;
  int ntile = kk >> 4;
  for (int t = 0; t < ntile; ++t) {
    int p0 = s * kk + (t << 4);
    // phase A: h = relu(WB @ x + bB) * attn  -> Hs[448][16] f16
    for (int rt = wave; rt < 28; rt += 8) {
      int ro = rt << 4;
      v8f acc = {};
      for (int k0 = 0; k0 < Kp; k0 += 32) {
        stage_A(Ast[wave], WB, Kp, ro, k0, lane);
        stage_B(Bst[wave], Xb, P, k0, p0, lane);
        asm volatile("s_wait_dscnt 0" ::: "memory");
        v16h a  = frag_A(Ast[wave], lane);
        v16h bb = frag_B(Bst[wave], lane);
        acc = __builtin_amdgcn_wmma_f32_16x16x32_f16(false, a, false, bb, (short)0, acc,
                                                     false, false);
        asm volatile("s_wait_dscnt 0" ::: "memory");
      }
#pragma unroll
      for (int i = 0; i < 8; ++i) {
        int o = ro + i + (hf << 3);
        float v = relu_f(acc[i] + biasB[o]) * wa[o];
        Hs[o * 16 + n] = (h16)v;
      }
    }
    __syncthreads();
    // phase B: fuse GEMM (cout x 448) reading B fragments straight from Hs
#pragma unroll
    for (int q = 0; q < 2; ++q) {
      int rt = wave + (q << 3);
      if (rt < Ot) {                   // wave-uniform branch, no barrier inside
        int ro = rt << 4;
        v8f acc = {};
        for (int k0 = 0; k0 < 448; k0 += 32) {
          stage_A(Ast[wave], WF, 448, ro, k0, lane);
          asm volatile("s_wait_dscnt 0" ::: "memory");
          v16h a  = frag_A(Ast[wave], lane);
          v16h bb = frag_B(Hs + k0 * 16, lane);
          acc = __builtin_amdgcn_wmma_f32_16x16x32_f16(false, a, false, bb, (short)0, acc,
                                                       false, false);
          asm volatile("s_wait_dscnt 0" ::: "memory");
        }
        float v[8];
#pragma unroll
        for (int i = 0; i < 8; ++i) v[i] = relu_f(acc[i] + biasF[ro + i + (hf << 3)]);
#pragma unroll
        for (int m = 1; m < 16; m <<= 1)
#pragma unroll
          for (int i = 0; i < 8; ++i) {
            float o2 = __shfl_xor(v[i], m, 32);
            v[i] = v[i] > o2 ? v[i] : o2;
          }
#pragma unroll
        for (int i = 0; i < 8; ++i) rm[q][i] = rm[q][i] > v[i] ? rm[q][i] : v[i];
      }
    }
    __syncthreads();
  }
  if (n == 0) {
#pragma unroll
    for (int q = 0; q < 2; ++q) {
      int rt = wave + (q << 3);
      if (rt < Ot) {
#pragma unroll
        for (int i = 0; i < 8; ++i) {
          int o = (rt << 4) + i + (hf << 3);
          out[((size_t)b * outRows + rowOff + o) * S + s] = rm[q][i];
        }
      }
    }
  }
}

// ---------------- scalar support kernels -------------------------------------
__global__ void transpose_xyz_kernel(const float* __restrict__ in, float* __restrict__ out,
                                     int N, int total)
{
  int t = blockIdx.x * blockDim.x + threadIdx.x;
  if (t >= total) return;
  int d = t % 3; int rem = t / 3; int n = rem % N; int b = rem / N;
  out[t] = in[((size_t)b * 3 + d) * N + n];
}

__global__ void fps_kernel(const float* __restrict__ xyz, int N, int S, int* __restrict__ idx)
{
  __shared__ float dist[4096];
  __shared__ float rv[256];
  __shared__ int   ri[256];
  int b = blockIdx.x, tid = threadIdx.x;
  const float* xb = xyz + (size_t)b * N * 3;
  for (int i = tid; i < N; i += 256) dist[i] = 1e10f;
  __syncthreads();
  int last = 0;
  for (int it = 0; it < S; ++it) {
    if (tid == 0) idx[b * S + it] = last;
    float lx = xb[last * 3], ly = xb[last * 3 + 1], lz = xb[last * 3 + 2];
    float bv = -1.f; int bi = 0;
    for (int i = tid; i < N; i += 256) {
      float dx = xb[i * 3] - lx, dy = xb[i * 3 + 1] - ly, dz = xb[i * 3 + 2] - lz;
      float d = dx * dx + dy * dy + dz * dz;
      float nd = dist[i] < d ? dist[i] : d;
      dist[i] = nd;
      if (nd > bv || (nd == bv && i < bi)) { bv = nd; bi = i; }
    }
    rv[tid] = bv; ri[tid] = bi;
    __syncthreads();
    for (int st = 128; st > 0; st >>= 1) {
      if (tid < st) {
        float v2 = rv[tid + st]; int i2 = ri[tid + st];
        if (v2 > rv[tid] || (v2 == rv[tid] && i2 < ri[tid])) { rv[tid] = v2; ri[tid] = i2; }
      }
      __syncthreads();
    }
    last = ri[0];
    __syncthreads();
  }
}

__global__ void gather_xyz_kernel(const float* __restrict__ src, const int* __restrict__ idx,
                                  float* __restrict__ dst, int S, int N, int total)
{
  int t = blockIdx.x * blockDim.x + threadIdx.x;
  if (t >= total) return;
  int b = t / S, s = t % S;
  int i = idx[b * S + s];
  dst[(size_t)(b * S + s) * 3 + 0] = src[((size_t)b * N + i) * 3 + 0];
  dst[(size_t)(b * S + s) * 3 + 1] = src[((size_t)b * N + i) * 3 + 1];
  dst[(size_t)(b * S + s) * 3 + 2] = src[((size_t)b * N + i) * 3 + 2];
}

__global__ void ball_group_kernel(const float* __restrict__ xyz, const float* __restrict__ nxyz,
                                  int N, int S, int k, float r2, int* __restrict__ grp, int total)
{
  int t = blockIdx.x * blockDim.x + threadIdx.x;
  if (t >= total) return;
  int b = t / S, s = t % S;
  const float* xb = xyz + (size_t)b * N * 3;
  float qx = nxyz[(size_t)(b * S + s) * 3 + 0];
  float qy = nxyz[(size_t)(b * S + s) * 3 + 1];
  float qz = nxyz[(size_t)(b * S + s) * 3 + 2];
  int* g = grp + (size_t)(b * S + s) * k;
  int cnt = 0;
  for (int i = 0; i < N && cnt < k; ++i) {
    float dx = xb[i * 3] - qx, dy = xb[i * 3 + 1] - qy, dz = xb[i * 3 + 2] - qz;
    if (dx * dx + dy * dy + dz * dz < r2) g[cnt++] = i;
  }
  int f = cnt > 0 ? g[0] : 0;
  for (int j = cnt; j < k; ++j) g[j] = f;
}

// CPF grouping: [feats(or rel), rel-c0, c0] -> X f16 [B][Kp][P], P = S*k
__global__ void build_x_kernel(const float* __restrict__ xyz, const float* __restrict__ nxyz,
                               const int* __restrict__ grp,
                               const float* __restrict__ feats, int featC, int featP,
                               h16* __restrict__ X, int P, int S, int k, int Kp, int total)
{
  int t = blockIdx.x * blockDim.x + threadIdx.x;
  if (t >= total) return;
  int j = t % k; int rem = t / k; int s = rem % S; int b = rem / S;
  int base = (b * S + s) * k;
  int i0 = grp[base], ii = grp[base + j];
  const float* xb = xyz + (size_t)b * featP * 3;   // featP == Nsrc
  float qx = nxyz[(size_t)(b * S + s) * 3 + 0];
  float qy = nxyz[(size_t)(b * S + s) * 3 + 1];
  float qz = nxyz[(size_t)(b * S + s) * 3 + 2];
  float rx = xb[ii * 3] - qx, ry = xb[ii * 3 + 1] - qy, rz = xb[ii * 3 + 2] - qz;
  float cx = xb[i0 * 3] - qx, cy = xb[i0 * 3 + 1] - qy, cz = xb[i0 * 3 + 2] - qz;
  size_t xo = (size_t)b * Kp * P + (size_t)(s * k + j);
  int cbase;
  if (feats) {
    const float* fb = feats + (size_t)b * featC * featP;
    for (int c = 0; c < featC; ++c) X[xo + (size_t)c * P] = (h16)fb[(size_t)c * featP + ii];
    cbase = featC;
  } else {
    X[xo + 0 * (size_t)P] = (h16)rx;
    X[xo + 1 * (size_t)P] = (h16)ry;
    X[xo + 2 * (size_t)P] = (h16)rz;
    cbase = 3;
  }
  X[xo + (size_t)(cbase + 0) * P] = (h16)(rx - cx);
  X[xo + (size_t)(cbase + 1) * P] = (h16)(ry - cy);
  X[xo + (size_t)(cbase + 2) * P] = (h16)(rz - cz);
  X[xo + (size_t)(cbase + 3) * P] = (h16)cx;
  X[xo + (size_t)(cbase + 4) * P] = (h16)cy;
  X[xo + (size_t)(cbase + 5) * P] = (h16)cz;
}

__global__ void pack_w_kernel(h16* __restrict__ dst, int Kp, int rowOff,
                              const float* __restrict__ src, int O, int Cin, int Orows)
{
  int t = blockIdx.x * blockDim.x + threadIdx.x;
  int total = Orows * Kp;
  if (t >= total) return;
  int o = t / Kp, c = t % Kp;
  float v = (o < O && c < Cin) ? src[o * Cin + c] : 0.f;
  dst[(size_t)(rowOff + o) * Kp + c] = (h16)v;
}

__global__ void pack_b_kernel(float* __restrict__ dst, int rowOff,
                              const float* __restrict__ src, int O, int Orows)
{
  int t = threadIdx.x;
  if (t >= Orows) return;
  dst[rowOff + t] = t < O ? src[t] : 0.f;
}

struct AttnArgs {
  const float* a1w[3]; const float* a1b[3];
  const float* a2w[3]; const float* a2b[3];
  float* Hsum; float* Watt; float invP;
};
__global__ void attention_kernel(AttnArgs A)
{
  __shared__ float mean[256];
  __shared__ float hid[32];
  int br = blockIdx.x, b = blockIdx.y, tid = threadIdx.x;
  const int osz[3] = {64, 128, 256};
  const int ofs[3] = {0, 64, 192};
  int o = osz[br], r = o >> 3, off = ofs[br];
  const float* hs = A.Hsum + b * 448 + off;
  for (int i = tid; i < o; i += 64) mean[i] = hs[i] * A.invP;
  __syncthreads();
  for (int j = tid; j < r; j += 64) {
    float s = A.a1b[br][j];
    for (int i = 0; i < o; ++i) s += A.a1w[br][j * o + i] * mean[i];
    hid[j] = relu_f(s);
  }
  __syncthreads();
  for (int i = tid; i < o; i += 64) {
    float s = A.a2b[br][i];
    for (int j = 0; j < r; ++j) s += A.a2w[br][i * r + j] * hid[j];
    A.Watt[b * 448 + off + i] = 1.f / (1.f + __expf(-s));
  }
}

__global__ void interp_kernel(const float* __restrict__ uxyz, int Nu,
                              const float* __restrict__ kxyz, int Nk,
                              const float* __restrict__ feats, int C, int featP,
                              h16* __restrict__ X, int Kp, int total)
{
  int t = blockIdx.x * blockDim.x + threadIdx.x;
  if (t >= total) return;
  int b = t / Nu, n = t % Nu;
  float qx = uxyz[((size_t)b * Nu + n) * 3 + 0];
  float qy = uxyz[((size_t)b * Nu + n) * 3 + 1];
  float qz = uxyz[((size_t)b * Nu + n) * 3 + 2];
  const float* kb = kxyz + (size_t)b * Nk * 3;
  float d0 = 1e30f, d1 = 1e30f, d2 = 1e30f;
  int i0 = 0, i1 = 0, i2 = 0;
  for (int m = 0; m < Nk; ++m) {
    float dx = kb[m * 3] - qx, dy = kb[m * 3 + 1] - qy, dz = kb[m * 3 + 2] - qz;
    float d = dx * dx + dy * dy + dz * dz;
    if (d < d0)      { d2 = d1; i2 = i1; d1 = d0; i1 = i0; d0 = d; i0 = m; }
    else if (d < d1) { d2 = d1; i2 = i1; d1 = d;  i1 = m; }
    else if (d < d2) { d2 = d;  i2 = m; }
  }
  float w0 = 1.f / (sqrtf(fmaxf(d0, 0.f)) + 1e-8f);
  float w1 = 1.f / (sqrtf(fmaxf(d1, 0.f)) + 1e-8f);
  float w2 = 1.f / (sqrtf(fmaxf(d2, 0.f)) + 1e-8f);
  float ws = w0 + w1 + w2;
  w0 /= ws; w1 /= ws; w2 /= ws;
  const float* fb = feats + (size_t)b * C * featP;
  for (int c = 0; c < C; ++c) {
    float v = w0 * fb[(size_t)c * featP + i0] + w1 * fb[(size_t)c * featP + i1] +
              w2 * fb[(size_t)c * featP + i2];
    X[((size_t)b * Kp + c) * Nu + n] = (h16)v;
  }
}

__global__ void copy_rows_kernel(h16* __restrict__ X, int Kp, int rowOff,
                                 const float* __restrict__ src, int C, int P, int total)
{
  int t = blockIdx.x * blockDim.x + threadIdx.x;
  if (t >= total) return;
  int p = t % P; int rem = t / P; int c = rem % C; int b = rem / C;
  X[((size_t)b * Kp + rowOff + c) * P + p] = (h16)src[((size_t)b * C + c) * P + p];
}

// ---------------- host orchestration -----------------------------------------
struct CamaP {
  const float *a1w[3], *a1b[3], *a2w[3], *a2b[3], *bw[3], *bb[3];
  const float *fuse_w, *fuse_b;
};

static CamaP parse_cama(void* const* d, int& i) {
  CamaP c;
  for (int br = 0; br < 3; ++br) {           // sorted keys: a1b,a1w,a2b,a2w,b,w
    c.a1b[br] = (const float*)d[i++]; c.a1w[br] = (const float*)d[i++];
    c.a2b[br] = (const float*)d[i++]; c.a2w[br] = (const float*)d[i++];
    c.bb[br]  = (const float*)d[i++]; c.bw[br]  = (const float*)d[i++];
  }
  c.fuse_b = (const float*)d[i++]; c.fuse_w = (const float*)d[i++];
  return c;
}

static void run_sa_scale(hipStream_t st,
                         const float* srcxyz, const float* newxyz, int Nsrc, int S, int k,
                         float radius, const float* feats, int featC,
                         const CamaP& cp, int cin, int Kp, int cout,
                         float* outF, int rowOff, int outRows,
                         int* grp, h16* Xbuf, h16* WB, float* BBp, h16* WF, float* BFp,
                         float* Hsum, float* Watt)
{
  int P = S * k;
  hipMemsetAsync(Xbuf, 0, (size_t)BATCH * Kp * P * sizeof(h16), st);
  int tot = BATCH * S;
  ball_group_kernel<<<(tot + 255) / 256, 256, 0, st>>>(srcxyz, newxyz, Nsrc, S, k,
                                                       radius * radius, grp, tot);
  int tot2 = BATCH * S * k;
  build_x_kernel<<<(tot2 + 255) / 256, 256, 0, st>>>(srcxyz, newxyz, grp, feats, featC,
                                                     Nsrc, Xbuf, P, S, k, Kp, tot2);
  const int osz[3] = {64, 128, 256}, ofs[3] = {0, 64, 192};
  for (int br = 0; br < 3; ++br) {
    int tw = osz[br] * Kp;
    pack_w_kernel<<<(tw + 255) / 256, 256, 0, st>>>(WB, Kp, ofs[br], cp.bw[br], osz[br],
                                                    cin, osz[br]);
    pack_b_kernel<<<1, 256, 0, st>>>(BBp, ofs[br], cp.bb[br], osz[br], osz[br]);
  }
  hipMemsetAsync(Hsum, 0, BATCH * 448 * sizeof(float), st);
  dim3 g1(P / 16, (448 + 63) / 64, BATCH);
  wmma_gemm_kernel<<<g1, 128, 0, st>>>(WB, BBp, Xbuf, Kp, P, 448, 448,
                                       nullptr, nullptr, Hsum);
  AttnArgs aa;
  for (int br = 0; br < 3; ++br) {
    aa.a1w[br] = cp.a1w[br]; aa.a1b[br] = cp.a1b[br];
    aa.a2w[br] = cp.a2w[br]; aa.a2b[br] = cp.a2b[br];
  }
  aa.Hsum = Hsum; aa.Watt = Watt; aa.invP = 1.f / (float)P;
  attention_kernel<<<dim3(3, BATCH), 64, 0, st>>>(aa);
  int tw = cout * 448;
  pack_w_kernel<<<(tw + 255) / 256, 256, 0, st>>>(WF, 448, 0, cp.fuse_w, cout, 448, cout);
  pack_b_kernel<<<1, 256, 0, st>>>(BFp, 0, cp.fuse_b, cout, cout);
  fuse_max_kernel<<<dim3(S, BATCH), 256, 0, st>>>(Xbuf, Kp, P, k, WB, BBp, Watt,
                                                  WF, BFp, cout, outF, rowOff, outRows, S);
}

extern "C" void kernel_launch(void* const* d_in, const int* in_sizes, int n_in,
                              void* d_out, int out_size, void* d_ws, size_t ws_size,
                              hipStream_t stream)
{
  (void)in_sizes; (void)out_size;
  if (n_in < 111) return;
  const float* xyz    = (const float*)d_in[0];
  int i = 1;                                    // sorted top-level param keys
  const float* fp1_b0 = (const float*)d_in[i++];
  const float* fp1_b1 = (const float*)d_in[i++];
  const float* fp1_w0 = (const float*)d_in[i++];
  const float* fp1_w1 = (const float*)d_in[i++];
  const float* fp2_b0 = (const float*)d_in[i++];
  const float* fp2_b1 = (const float*)d_in[i++];
  const float* fp2_b2 = (const float*)d_in[i++];
  const float* fp2_w0 = (const float*)d_in[i++];
  const float* fp2_w1 = (const float*)d_in[i++];
  const float* fp2_w2 = (const float*)d_in[i++];
  CamaP sa1[3], sa2[2];
  for (int s = 0; s < 3; ++s) sa1[s] = parse_cama(d_in, i);
  for (int s = 0; s < 2; ++s) sa2[s] = parse_cama(d_in, i);

  uint8_t* wsp = (uint8_t*)d_ws;
  size_t off = 0;
  auto alloc = [&](size_t bytes) -> void* {
    void* p = wsp + off;
    off += (bytes + 255) & ~(size_t)255;
    return p;
  };
  float* xyzt   = (float*)alloc((size_t)BATCH * NPTS * 3 * 4);
  float* l1xyz  = (float*)alloc((size_t)BATCH * 1024 * 3 * 4);
  float* l2xyz  = (float*)alloc((size_t)BATCH * 256 * 3 * 4);
  int*   fpsidx = (int*)  alloc((size_t)BATCH * 1024 * 4);
  int*   grp    = (int*)  alloc((size_t)BATCH * 1024 * 64 * 4);
  float* Hsum   = (float*)alloc((size_t)BATCH * 448 * 4);
  float* Watt   = (float*)alloc((size_t)BATCH * 448 * 4);
  h16*   WB     = (h16*)  alloc(512 * 1024);
  h16*   WF     = (h16*)  alloc(512 * 1024);
  float* BBp    = (float*)alloc(2048);
  float* BFp    = (float*)alloc(2048);
  float* l1f    = (float*)alloc((size_t)BATCH * 320 * 1024 * 4);
  float* l2f    = (float*)alloc((size_t)BATCH * 512 * 256 * 4);
  float* fp1o   = (float*)alloc((size_t)BATCH * 256 * 1024 * 4);
  h16*   X2A    = (h16*)  alloc((size_t)BATCH * 256 * 4096 * 2);
  h16*   X2B    = (h16*)  alloc((size_t)BATCH * 256 * 4096 * 2);
  h16*   Xbuf   = (h16*)  alloc((size_t)BATCH * 352 * 16384 * 2);
  if (off > ws_size) return;

  int tt = BATCH * 3 * NPTS;
  transpose_xyz_kernel<<<(tt + 255) / 256, 256, 0, stream>>>(xyz, xyzt, NPTS, tt);
  fps_kernel<<<BATCH, 256, 0, stream>>>(xyzt, NPTS, 1024, fpsidx);
  { int g = BATCH * 1024;
    gather_xyz_kernel<<<(g + 255) / 256, 256, 0, stream>>>(xyzt, fpsidx, l1xyz, 1024, NPTS, g); }

  // ---- SA1 (cin 9 -> Kp 32), couts 64/128/128 -> l1f rows 0/64/192, 320 total
  const float r1[3] = {0.1f, 0.2f, 0.4f};
  const int   k1[3] = {16, 32, 64};
  const int  co1[3] = {64, 128, 128};
  const int  ro1[3] = {0, 64, 192};
  for (int sc = 0; sc < 3; ++sc)
    run_sa_scale(stream, xyzt, l1xyz, NPTS, 1024, k1[sc], r1[sc], nullptr, 0,
                 sa1[sc], 9, 32, co1[sc], l1f, ro1[sc], 320,
                 grp, Xbuf, WB, BBp, WF, BFp, Hsum, Watt);

  // ---- SA2 (cin 326 -> Kp 352), couts 256/256 -> l2f rows 0/256, 512 total
  fps_kernel<<<BATCH, 256, 0, stream>>>(l1xyz, 1024, 256, fpsidx);
  { int g = BATCH * 256;
    gather_xyz_kernel<<<(g + 255) / 256, 256, 0, stream>>>(l1xyz, fpsidx, l2xyz, 256, 1024, g); }
  const float r2a[2] = {0.4f, 0.8f};
  const int   k2[2] = {32, 64};
  const int  ro2[2] = {0, 256};
  for (int sc = 0; sc < 2; ++sc)
    run_sa_scale(stream, l1xyz, l2xyz, 1024, 256, k2[sc], r2a[sc], l1f, 320,
                 sa2[sc], 326, 352, 256, l2f, ro2[sc], 512,
                 grp, Xbuf, WB, BBp, WF, BFp, Hsum, Watt);

  // ---- FP1: interp(512) + l1f(320) = 832 -> 256 -> 256
  { int g = BATCH * 1024;
    interp_kernel<<<(g + 255) / 256, 256, 0, stream>>>(l1xyz, 1024, l2xyz, 256,
                                                       l2f, 512, 256, Xbuf, 832, g); }
  { int g = BATCH * 320 * 1024;
    copy_rows_kernel<<<(g + 255) / 256, 256, 0, stream>>>(Xbuf, 832, 512, l1f, 320, 1024, g); }
  { int tw = 256 * 832;
    pack_w_kernel<<<(tw + 255) / 256, 256, 0, stream>>>(WB, 832, 0, fp1_w0, 256, 832, 256);
    pack_b_kernel<<<1, 256, 0, stream>>>(BBp, 0, fp1_b0, 256, 256);
    dim3 g(1024 / 16, 4, BATCH);
    wmma_gemm_kernel<<<g, 128, 0, stream>>>(WB, BBp, Xbuf, 832, 1024, 256, 256,
                                            nullptr, X2A, nullptr); }
  { int tw = 256 * 256;
    pack_w_kernel<<<(tw + 255) / 256, 256, 0, stream>>>(WF, 256, 0, fp1_w1, 256, 256, 256);
    pack_b_kernel<<<1, 256, 0, stream>>>(BFp, 0, fp1_b1, 256, 256);
    dim3 g(1024 / 16, 4, BATCH);
    wmma_gemm_kernel<<<g, 128, 0, stream>>>(WF, BFp, X2A, 256, 1024, 256, 256,
                                            fp1o, nullptr, nullptr); }

  // ---- FP2: interp(256) -> 256 -> 128 -> 2 (relu on all layers, incl. last)
  { int g = BATCH * 4096;
    interp_kernel<<<(g + 255) / 256, 256, 0, stream>>>(xyzt, 4096, l1xyz, 1024,
                                                       fp1o, 256, 1024, Xbuf, 256, g); }
  { int tw = 256 * 256;
    pack_w_kernel<<<(tw + 255) / 256, 256, 0, stream>>>(WB, 256, 0, fp2_w0, 256, 256, 256);
    pack_b_kernel<<<1, 256, 0, stream>>>(BBp, 0, fp2_b0, 256, 256);
    dim3 g(4096 / 16, 4, BATCH);
    wmma_gemm_kernel<<<g, 128, 0, stream>>>(WB, BBp, Xbuf, 256, 4096, 256, 256,
                                            nullptr, X2A, nullptr); }
  { int tw = 128 * 256;
    pack_w_kernel<<<(tw + 255) / 256, 256, 0, stream>>>(WF, 256, 0, fp2_w1, 128, 256, 128);
    pack_b_kernel<<<1, 256, 0, stream>>>(BFp, 0, fp2_b1, 128, 128);
    dim3 g(4096 / 16, 2, BATCH);
    wmma_gemm_kernel<<<g, 128, 0, stream>>>(WF, BFp, X2A, 256, 4096, 128, 128,
                                            nullptr, X2B, nullptr); }
  { int tw = 16 * 128;
    pack_w_kernel<<<(tw + 255) / 256, 256, 0, stream>>>(WB, 128, 0, fp2_w2, 2, 128, 16);
    pack_b_kernel<<<1, 256, 0, stream>>>(BBp, 0, fp2_b2, 2, 16);
    dim3 g(4096 / 16, 1, BATCH);
    wmma_gemm_kernel<<<g, 128, 0, stream>>>(WB, BBp, X2B, 128, 4096, 16, 2,
                                            (float*)d_out, nullptr, nullptr); }
}
